// FrozenMLIPEncoder_80290118631445
// MI455X (gfx1250) — matrix operand order repaired and checked by reading, not compiled, not run on hardware
//
#include <hip/hip_runtime.h>
#include <hip/hip_bf16.h>

typedef __attribute__((ext_vector_type(16))) _Float16 v16h;
typedef __attribute__((ext_vector_type(8)))  float    v8f;

#define BATCH 256
#define NATOM 64
#define EMBD  64
#define NG    64
#define HID   128
#define OUTD  512
#define LDH   132   // padded f16 row stride (132*2B = 264B = 66 banks -> conflict-free frags)

// ---- WMMA fragment helpers (wave32; contiguous k, not the table order of CDNA5 ISA 7.12.2) ----
// 16-bit A 16x32, this code's contiguous k order (same as B): lanes 0-15 row M=lane, halfs0-15 = K0..15;
//                 lanes 16-31 same rows, K16..31.
__device__ __forceinline__ v16h load_a_frag(const _Float16* base, int ld, int row0,
                                            int k0, int lane) {
    const int r  = lane & 15;
    const int hi = lane >> 4;
    const _Float16* p = base + (row0 + r) * ld + k0;
    v16h a;
#pragma unroll
    for (int i = 0; i < 16; ++i) a[i] = p[hi * 16 + i];
    return a;
}

// 16-bit B 32x16 from a TRANSPOSED weight tile Wt[col][k] (contiguous K per lane):
// lanes 0-15 col N=lane hold K0..15; lanes 16-31 hold K16..31.
__device__ __forceinline__ v16h load_b_frag_t(const _Float16* base, int ld, int col0,
                                              int k0, int lane) {
    const int c  = lane & 15;
    const int hi = lane >> 4;
    const _Float16* p = base + (col0 + c) * ld + k0 + hi * 16;
    v16h b;
#pragma unroll
    for (int i = 0; i < 16; ++i) b[i] = p[i];
    return b;
}

__device__ __forceinline__ v8f wmma_f16(v16h a, v16h b, v8f c) {
    // (neg_a, A, neg_b, B, c_mod, C, reuse_a, reuse_b)
    return __builtin_amdgcn_wmma_f32_16x16x32_f16(false, a, false, b, (short)0, c,
                                                  false, false);
}

__device__ __forceinline__ float gelu_exact(float x) {
    return 0.5f * x * (1.0f + erff(x * 0.70710678118654752f));
}

// =====================================================================
// Kernel 1: one block per molecule. Featurize (dist -> gaussian smear ->
// embed gather) entirely in LDS, then feat@W1 -> GELU -> @W2 via WMMA,
// then mean over atoms -> mol[HID] (f16) to workspace.
// =====================================================================
__global__ __launch_bounds__(256) void mlip_encode_kernel(
    const int*   __restrict__ atomic_nums,   // [B,N]
    const float* __restrict__ coords,        // [B,N,3]
    const float* __restrict__ embed_table,   // [120,EMB]
    const float* __restrict__ W1,            // [EMB+G, HID] (in,out)
    const float* __restrict__ b1,            // [HID]
    const float* __restrict__ W2,            // [HID, HID]
    const float* __restrict__ b2,            // [HID]
    const float* __restrict__ centers,       // [G]
    _Float16*    __restrict__ molOut)        // [B, HID] f16
{
    __shared__ __align__(16) _Float16 sFeat[NATOM][LDH];  // 16.9 KB, feat then h1
    __shared__ __align__(16) _Float16 sW[HID][LDH];       // 33.8 KB, W1^T / W2^T (dist reuses it)
    __shared__ float sCoords[NATOM * 3];
    __shared__ float sMol[HID];

    const int b    = blockIdx.x;
    const int t    = threadIdx.x;
    const int lane = t & 31;
    const int wave = t >> 5;

    // ---- Phase A: coordinates ----
    if (t < NATOM * 3) sCoords[t] = coords[b * NATOM * 3 + t];
    __syncthreads();

    // pairwise distances into the (not-yet-used) weight buffer
    float* dist = (float*)&sW[0][0];  // 64*64*4 = 16 KB <= 33.8 KB
    {
        const int p0 = t * 16;
#pragma unroll
        for (int q = 0; q < 16; ++q) {
            const int p = p0 + q;
            const int i = p >> 6, j = p & 63;
            const float dx = sCoords[i * 3 + 0] - sCoords[j * 3 + 0];
            const float dy = sCoords[i * 3 + 1] - sCoords[j * 3 + 1];
            const float dz = sCoords[i * 3 + 2] - sCoords[j * 3 + 2];
            const float sq = dx * dx + dy * dy + dz * dz;
            dist[p] = sq > 0.0f ? sqrtf(sq) : 0.0f;
        }
    }
    __syncthreads();

    // ---- Gaussian smearing mean over neighbors + embedding gather ----
    {
        const int i   = t >> 2;         // atom
        const int seg = (t & 3) * 16;   // 16-wide gaussian / embed segment
        float cg[16], acc[16];
#pragma unroll
        for (int g = 0; g < 16; ++g) { cg[g] = centers[seg + g]; acc[g] = 0.0f; }
        for (int j = 0; j < NATOM; ++j) {
            const float d = dist[i * NATOM + j];
#pragma unroll
            for (int g = 0; g < 16; ++g) {
                const float u = (d - cg[g]) * 2.0f;          // / GAUSS_WIDTH(0.5)
                acc[g] += __expf(-0.5f * u * u);             // v_exp_f32 trans pipe
            }
        }
        const int an = atomic_nums[b * NATOM + i];
        const float* er = embed_table + an * EMBD + seg;
#pragma unroll
        for (int g = 0; g < 16; ++g) {
            sFeat[i][seg + g]        = (_Float16)er[g];
            sFeat[i][EMBD + seg + g] = (_Float16)(acc[g] * (1.0f / NATOM));
        }
    }
    __syncthreads();  // dist reads done; safe to overwrite sW

    // ---- W1^T into LDS (coalesced global reads, f32->f16) ----
    for (int e = t; e < HID * HID; e += 256) {
        const int o = e & 127, k = e >> 7;
        sW[o][k] = (_Float16)W1[k * HID + o];
    }
    if (t < HID) sMol[t] = 0.0f;
    __syncthreads();

    // ---- GEMM1: feat[64,128] @ W1 -> 4x8 tile grid, 4 tiles/wave ----
    const int mt  = wave & 3;
    const int nt0 = (wave >> 2) * 4;
    v8f acc1[4] = {};
    for (int kc = 0; kc < 4; ++kc) {
        const v16h a = load_a_frag(&sFeat[0][0], LDH, mt * 16, kc * 32, lane);
#pragma unroll
        for (int i = 0; i < 4; ++i) {
            const v16h bb = load_b_frag_t(&sW[0][0], LDH, (nt0 + i) * 16, kc * 32, lane);
            acc1[i] = wmma_f16(a, bb, acc1[i]);
        }
    }
    __syncthreads();  // all waves done reading sFeat/sW

    // ---- bias + exact GELU, write h1 back in place over sFeat (f16) ----
    {
        const int cn = lane & 15;
        const int hi = lane >> 4;
#pragma unroll
        for (int i = 0; i < 4; ++i) {
            const int n = (nt0 + i) * 16 + cn;
            const float bias = b1[n];
#pragma unroll
            for (int r = 0; r < 8; ++r) {
                const int m = mt * 16 + r + 8 * hi;   // C/D layout: VGPR r -> row r (+8 for hi lanes)
                sFeat[m][n] = (_Float16)gelu_exact(acc1[i][r] + bias);
            }
        }
    }
    // ---- W2^T into LDS ----
    for (int e = t; e < HID * HID; e += 256) {
        const int o = e & 127, k = e >> 7;
        sW[o][k] = (_Float16)W2[k * HID + o];
    }
    __syncthreads();

    // ---- GEMM2: h1[64,128] @ W2, fold atom-mean directly from fragments ----
    v8f acc2[4] = {};
    for (int kc = 0; kc < 4; ++kc) {
        const v16h a = load_a_frag(&sFeat[0][0], LDH, mt * 16, kc * 32, lane);
#pragma unroll
        for (int i = 0; i < 4; ++i) {
            const v16h bb = load_b_frag_t(&sW[0][0], LDH, (nt0 + i) * 16, kc * 32, lane);
            acc2[i] = wmma_f16(a, bb, acc2[i]);
        }
    }
    {
        const int cn = lane & 15;
#pragma unroll
        for (int i = 0; i < 4; ++i) {
            float ps = 0.0f;
#pragma unroll
            for (int r = 0; r < 8; ++r) ps += acc2[i][r];   // 8 rows of one column
            atomicAdd(&sMol[(nt0 + i) * 16 + cn], ps);      // ds_add_f32
        }
    }
    __syncthreads();
    if (t < HID)
        molOut[b * HID + t] = (_Float16)(sMol[t] * (1.0f / NATOM) + b2[t]);
}

// =====================================================================
// Kernel 2: out[256,512] = mol[256,128] @ Wo + bo.
// 64 blocks: blockIdx = (mtile 0..15) + 16*(nchunk 0..3); 8 waves, one
// 16x16 WMMA tile each, K-loop of 4.
// =====================================================================
__global__ __launch_bounds__(256) void mlip_proj_kernel(
    const _Float16* __restrict__ mol,   // [B, HID] f16
    const float*    __restrict__ Wo,    // [HID, OUT]
    const float*    __restrict__ bo,    // [OUT]
    float*          __restrict__ out)   // [B, OUT]
{
    __shared__ __align__(16) _Float16 sA[16][LDH];
    __shared__ __align__(16) _Float16 sW[128][LDH];

    const int t     = threadIdx.x;
    const int lane  = t & 31;
    const int wave  = t >> 5;
    const int mtile = blockIdx.x & 15;
    const int n0    = (blockIdx.x >> 4) * 128;

    for (int e = t; e < 16 * HID; e += 256) {
        const int r = e >> 7, k = e & 127;
        sA[r][k] = mol[(mtile * 16 + r) * HID + k];
    }
    for (int e = t; e < 128 * 128; e += 256) {   // coalesced over columns
        const int c = e & 127, k = e >> 7;
        sW[c][k] = (_Float16)Wo[k * OUTD + n0 + c];
    }
    __syncthreads();

    v8f acc = {};
    for (int kc = 0; kc < 4; ++kc) {
        const v16h a  = load_a_frag(&sA[0][0], LDH, 0, kc * 32, lane);
        const v16h bb = load_b_frag_t(&sW[0][0], LDH, wave * 16, kc * 32, lane);
        acc = wmma_f16(a, bb, acc);
    }
    const int cn  = lane & 15;
    const int hi  = lane >> 4;
    const int col = n0 + wave * 16 + cn;
    const float bias = bo[col];
#pragma unroll
    for (int r = 0; r < 8; ++r) {
        const int row = mtile * 16 + r + 8 * hi;
        out[row * OUTD + col] = acc[r] + bias;
    }
}

extern "C" void kernel_launch(void* const* d_in, const int* in_sizes, int n_in,
                              void* d_out, int out_size, void* d_ws, size_t ws_size,
                              hipStream_t stream) {
    const int*   atomic_nums = (const int*)  d_in[0];
    const float* coords      = (const float*)d_in[1];
    const float* embed_table = (const float*)d_in[2];
    const float* W1          = (const float*)d_in[3];
    const float* b1          = (const float*)d_in[4];
    const float* W2          = (const float*)d_in[5];
    const float* b2          = (const float*)d_in[6];
    const float* Wo          = (const float*)d_in[7];
    const float* bo          = (const float*)d_in[8];
    const float* centers     = (const float*)d_in[9];
    float*       out         = (float*)d_out;

    _Float16* mol = (_Float16*)d_ws;   // 256*128*2B = 64 KB scratch

    mlip_encode_kernel<<<BATCH, 256, 0, stream>>>(atomic_nums, coords, embed_table,
                                                  W1, b1, W2, b2, centers, mol);
    mlip_proj_kernel<<<64, 256, 0, stream>>>(mol, Wo, bo, out);
}
